// PairContrastiveLoss_69174743269979
// MI455X (gfx1250) — matrix operand order repaired and checked by reading, not compiled, not run on hardware
//
#include <hip/hip_runtime.h>
#include <hip/hip_bf16.h>

typedef __attribute__((ext_vector_type(16))) __bf16 v16bf;
typedef __attribute__((ext_vector_type(8)))  __bf16 v8bf;
typedef __attribute__((ext_vector_type(8)))  float  v8f;

// ---------------------------------------------------------------------------
// Kernel 0: convert weights f32 -> bf16 (layout already [n][k], k=c*256+ky*16+kx)
//           and zero the 64x64 CE accumulator.
// ---------------------------------------------------------------------------
__global__ void prep_kernel(const float* __restrict__ W, __bf16* __restrict__ Wt,
                            float* __restrict__ G) {
  int idx = blockIdx.x * 256 + threadIdx.x;
  if (idx < 294912) Wt[idx] = (__bf16)W[idx];
  if (idx < 4096)   G[idx] = 0.0f;
}

// ---------------------------------------------------------------------------
// A-fragment (16x32 bf16) gathered directly from NCHW f32 image (fused im2col).
// K order used here (contiguous): lanes 0-15 row=lane, K in {0..15}; lanes 16-31 row=lane-16,
// K in {16..31}. Each 8-wide K chunk is contiguous in kx -> 2x float4.
// ---------------------------------------------------------------------------
__device__ inline v16bf load_a_patch(const float* __restrict__ abase, int kbase, int lane) {
  int h0 = kbase + ((lane & 16) ? 16 : 0);
  int h1 = kbase + 8 + ((lane & 16) ? 16 : 0);
  v16bf a;
  {
    int c = h0 >> 8, rem = h0 & 255;
    const float* p = abase + c * 50176 + (rem >> 4) * 224 + (rem & 15);
    float4 f0 = *(const float4*)p;
    float4 f1 = *(const float4*)(p + 4);
    a[0] = (__bf16)f0.x; a[1] = (__bf16)f0.y; a[2] = (__bf16)f0.z; a[3] = (__bf16)f0.w;
    a[4] = (__bf16)f1.x; a[5] = (__bf16)f1.y; a[6] = (__bf16)f1.z; a[7] = (__bf16)f1.w;
  }
  {
    int c = h1 >> 8, rem = h1 & 255;
    const float* p = abase + c * 50176 + (rem >> 4) * 224 + (rem & 15);
    float4 f0 = *(const float4*)p;
    float4 f1 = *(const float4*)(p + 4);
    a[8]  = (__bf16)f0.x; a[9]  = (__bf16)f0.y; a[10] = (__bf16)f0.z; a[11] = (__bf16)f0.w;
    a[12] = (__bf16)f1.x; a[13] = (__bf16)f1.y; a[14] = (__bf16)f1.z; a[15] = (__bf16)f1.w;
  }
  return a;
}

// ---------------------------------------------------------------------------
// Kernel 1: fused patch-embed GEMM (12544x768 @ 768x384) + bias + softmax /
// log-softmax, writing bf16 results in [image][k = patch*384 + c] layout.
// One wave = 16 rows x 384 cols (24 f32 accumulators). grid (196, 2), block 128.
// ---------------------------------------------------------------------------
__global__ __launch_bounds__(128)
void patch_gemm_softmax(const float* __restrict__ x, const float* __restrict__ y,
                        const __bf16* __restrict__ Wt, const float* __restrict__ bias,
                        __bf16* __restrict__ Lx, __bf16* __restrict__ Qy) {
  const int lane    = threadIdx.x & 31;
  const int wave    = threadIdx.x >> 5;
  const int rowbase = blockIdx.x * 64 + wave * 16;
  const bool isY    = (blockIdx.y != 0);
  const float* img  = isY ? y : x;
  __bf16* outp      = isY ? Qy : Lx;

  // per-lane A row -> patch coordinates
  const int m  = rowbase + (lane & 15);
  const int b  = m / 196;
  const int p  = m % 196;
  const int ph = p / 14, pw = p % 14;
  const float* abase = img + (size_t)b * 150528 + (ph * 16) * 224 + pw * 16;

  const int nlo  = lane & 15;
  const int koff = (lane & 16) ? 16 : 0;

  v8f acc[24];
#pragma unroll
  for (int t = 0; t < 24; ++t) acc[t] = (v8f)(0.0f);

  for (int ks = 0; ks < 24; ++ks) {
    v16bf afrag = load_a_patch(abase, ks * 32, lane);
    const __bf16* wp = Wt + koff + ks * 32;
#pragma unroll
    for (int t = 0; t < 24; ++t) {
      v16bf bfrag = *(const v16bf*)(wp + (size_t)(t * 16 + nlo) * 768);
      acc[t] = __builtin_amdgcn_wmma_f32_16x16x32_bf16(
          false, afrag, false, bfrag, (short)0, acc[t], false, false);
    }
  }

  // bias (all 8 rows in a VGPR share column n = t*16 + nlo)
#pragma unroll
  for (int t = 0; t < 24; ++t) {
    float bv = bias[t * 16 + nlo];
#pragma unroll
    for (int r = 0; r < 8; ++r) acc[t][r] += bv;
  }

  // softmax statistics per row; rows are per-VGPR r, shared across the 16 lanes
  // of each half-group -> shfl_xor masks 1,2,4,8 stay within the half (wave32).
  float rmax[8], rsum[8];
#pragma unroll
  for (int r = 0; r < 8; ++r) {
    float mx = acc[0][r];
#pragma unroll
    for (int t = 1; t < 24; ++t) mx = fmaxf(mx, acc[t][r]);
    for (int s = 1; s < 16; s <<= 1) mx = fmaxf(mx, __shfl_xor(mx, s, 32));
    rmax[r] = mx;
    float sum = 0.0f;
#pragma unroll
    for (int t = 0; t < 24; ++t) sum += __expf(acc[t][r] - mx);
    for (int s = 1; s < 16; s <<= 1) sum += __shfl_xor(sum, s, 32);
    rsum[r] = sum;
  }

  // write log-softmax (x) / softmax (y) as bf16; flat index (row*384 + n)
  // == [image b][k = patch*384 + c] which is exactly the GEMM-2 layout.
#pragma unroll
  for (int r = 0; r < 8; ++r) {
    int mr = rowbase + r + ((lane & 16) ? 8 : 0);
    __bf16* op = outp + (size_t)mr * 384 + nlo;
    if (!isY) {
      float lse = rmax[r] + __logf(rsum[r]);
#pragma unroll
      for (int t = 0; t < 24; ++t) op[t * 16] = (__bf16)(acc[t][r] - lse);
    } else {
      float inv = 1.0f / rsum[r];
#pragma unroll
      for (int t = 0; t < 24; ++t) op[t * 16] = (__bf16)(__expf(acc[t][r] - rmax[r]) * inv);
    }
  }
}

// ---------------------------------------------------------------------------
// Kernel 2: G[i][j] = sum_k Lx[i][k] * Qy[j][k]  (64x64, K = 75264)
// split-K over 196 patches (K-chunk 384 = 12 WMMA steps), f32 atomics into G.
// 784 blocks x 4 waves = 3136 waves = 16 tiles * 196 splits.
// ---------------------------------------------------------------------------
__global__ __launch_bounds__(128)
void ce_gemm(const __bf16* __restrict__ Lx, const __bf16* __restrict__ Qy,
             float* __restrict__ G) {
  const int lane  = threadIdx.x & 31;
  const int wid   = blockIdx.x * 4 + (threadIdx.x >> 5);
  const int tile  = wid & 15;
  const int split = wid >> 4;            // 0..195
  const int ti    = (tile >> 2) * 16;
  const int tj    = (tile & 3) * 16;
  const int nlo   = lane & 15;
  const size_t K  = 75264;

  const __bf16* arow = Lx + (size_t)(ti + nlo) * K + (size_t)split * 384;
  const __bf16* brow = Qy + (size_t)(tj + nlo) * K + (size_t)split * 384
                          + ((lane & 16) ? 16 : 0);
  const int h0 = (lane & 16) ? 16 : 0;

  v8f acc = (v8f)(0.0f);
  for (int ks = 0; ks < 12; ++ks) {
    int kb = ks * 32;
    v8bf lo = *(const v8bf*)(arow + kb + h0);
    v8bf hi = *(const v8bf*)(arow + kb + 8 + h0);
    v16bf a;
#pragma unroll
    for (int i = 0; i < 8; ++i) { a[i] = lo[i]; a[8 + i] = hi[i]; }
    v16bf bfrag = *(const v16bf*)(brow + kb);
    acc = __builtin_amdgcn_wmma_f32_16x16x32_bf16(
        false, a, false, bfrag, (short)0, acc, false, false);
  }
#pragma unroll
  for (int r = 0; r < 8; ++r) {
    int gi = ti + r + ((lane & 16) ? 8 : 0);
    atomicAdd(&G[gi * 64 + tj + nlo], acc[r]);
  }
}

// ---------------------------------------------------------------------------
// Kernel 3: scalar loss. CE = -G/196; close/far reduces to 63*tr / (sum - tr).
// ---------------------------------------------------------------------------
__global__ void finalize_kernel(const float* __restrict__ G, float* __restrict__ out) {
  __shared__ float ssum[64];
  __shared__ float sdia[64];
  int i = threadIdx.x;
  float s = 0.0f;
  for (int j = 0; j < 64; ++j) s += G[i * 64 + j];
  ssum[i] = s;
  sdia[i] = G[i * 64 + i];
  __syncthreads();
  if (i == 0) {
    float tot = 0.0f, tr = 0.0f;
    for (int j = 0; j < 64; ++j) { tot += ssum[j]; tr += sdia[j]; }
    out[0] = 63.0f * tr / (tot - tr);
  }
}

// ---------------------------------------------------------------------------
extern "C" void kernel_launch(void* const* d_in, const int* in_sizes, int n_in,
                              void* d_out, int out_size, void* d_ws, size_t ws_size,
                              hipStream_t stream) {
  const float* x    = (const float*)d_in[0];   // (64,3,224,224) f32
  const float* y    = (const float*)d_in[1];   // (64,3,224,224) f32
  const float* W    = (const float*)d_in[2];   // (384,3,16,16)  f32
  const float* bias = (const float*)d_in[3];   // (384,)         f32

  char* ws = (char*)d_ws;
  __bf16* Wt = (__bf16*)ws;                              // 589,824 B
  __bf16* Lx = (__bf16*)(ws + 655360);                   // 9,633,792 B
  __bf16* Qy = (__bf16*)(ws + 655360 + 9633792);         // 9,633,792 B
  float*  G  = (float*)(ws + 655360 + 2 * 9633792);      // 16,384 B (~20 MB total)
  float* out = (float*)d_out;

  prep_kernel<<<1152, 256, 0, stream>>>(W, Wt, G);
  patch_gemm_softmax<<<dim3(196, 2), 128, 0, stream>>>(x, y, Wt, bias, Lx, Qy);
  ce_gemm<<<784, 128, 0, stream>>>(Lx, Qy, G);
  finalize_kernel<<<1, 64, 0, stream>>>(G, out);
}
